// multihead_self_attention_53223234732522
// MI455X (gfx1250) — compile-verified, matrix operand order checked by reading, not run on hardware
//
#include <hip/hip_runtime.h>
#include <hip/hip_bf16.h>

typedef __attribute__((ext_vector_type(16))) _Float16 v16h;
typedef __attribute__((ext_vector_type(8)))  _Float16 v8h;
typedef __attribute__((ext_vector_type(8)))  float    v8f;

#define HH 16      // heads
#define BB 2       // batch
#define SS 2048    // sequence
#define DD 1024    // model dim
#define DH 64      // head dim

// Types for the async-to-LDS builtin, matching hipcc's reported signature:
//   (int_vec16* /*generic*/, int_vec16 __shared__*, Iint offset, Iint cpol)
typedef int vsi4 __attribute__((vector_size(16)));
typedef __attribute__((address_space(3))) vsi4 svsi4;  // LDS int4

// ---------------------------------------------------------------------------
// Async global->LDS 16-byte copy (CDNA5 GLOBAL_LOAD_ASYNC_TO_LDS_B128,
// tracked by ASYNCcnt). Guarded: falls back to a synchronous copy if the
// builtin is unavailable, so the file always compiles.
// ---------------------------------------------------------------------------
__device__ __forceinline__ void async_copy16(const _Float16* g, _Float16* l) {
#if defined(__HIP_DEVICE_COMPILE__) && __has_builtin(__builtin_amdgcn_global_load_async_to_lds_b128)
    __builtin_amdgcn_global_load_async_to_lds_b128((vsi4*)g, (svsi4*)l, 0, 0);
#else
    *(v8h*)l = *(const v8h*)g;
#endif
}

__device__ __forceinline__ void wait_async_le(int n) {
#if defined(__HIP_DEVICE_COMPILE__) && __has_builtin(__builtin_amdgcn_s_wait_asynccnt)
    if (n == 0) __builtin_amdgcn_s_wait_asynccnt(0);
    else        __builtin_amdgcn_s_wait_asynccnt(16);
#else
    if (n == 0) asm volatile("s_wait_asynccnt 0" ::: "memory");
    else        asm volatile("s_wait_asynccnt 16" ::: "memory");
#endif
    asm volatile("" ::: "memory");   // keep LDS reads below the wait
}

// ---------------------------------------------------------------------------
// Load a 16xK=32 f16 WMMA A/B operand from a row-major matrix whose
// contraction dimension is contiguous (row stride = ld f16 elements).
// Per the CDNA5 16-bit A/B layout this is exactly two 16B loads per lane:
//   lane L: row = L&15, half = L>>4
//   v16h[0..7]  = row elements [half*8    .. half*8+7 ]
//   v16h[8..15] = row elements [16+half*8 .. 16+half*8+7]
// Works for both global (global_load_b128) and LDS (ds_load_b128) pointers.
// ---------------------------------------------------------------------------
__device__ __forceinline__ v16h load_ab16x32(const _Float16* __restrict__ base,
                                             int ld, int lane) {
    const int row  = lane & 15;
    const int half = lane >> 4;
    const _Float16* p = base + (size_t)row * (size_t)ld + half * 8;
    const v8h lo = *(const v8h*)(p);
    const v8h hi = *(const v8h*)(p + 16);
    v16h r;
#pragma unroll
    for (int i = 0; i < 8; ++i) { r[i] = lo[i]; r[i + 8] = hi[i]; }
    return r;
}

// ---------------------------------------------------------------------------
// fp32 -> fp16 convert
// ---------------------------------------------------------------------------
__global__ void mha_cvt_f16(const float* __restrict__ s, _Float16* __restrict__ d, int n) {
    int i = blockIdx.x * blockDim.x + threadIdx.x;
    if (i < n) d[i] = (_Float16)s[i];
}

// ---------------------------------------------------------------------------
// C[M,N] (f32, row-major) = A[M,K] (f16 rm) * B[N,K]^T (f16 rm)
// One wave per 32x64 C block: 2 A-tiles x 4 B-tiles, K-step 32.
// ---------------------------------------------------------------------------
__global__ __launch_bounds__(32)
void mha_gemm_tn(const _Float16* __restrict__ A, const _Float16* __restrict__ B,
                 float* __restrict__ C, int M, int N, int K) {
    const int lane = threadIdx.x;
    const int n0 = blockIdx.x * 64;
    const int m0 = blockIdx.y * 32;

    v8f acc[2][4];
#pragma unroll
    for (int i = 0; i < 2; ++i)
#pragma unroll
        for (int j = 0; j < 4; ++j) acc[i][j] = {};

    for (int kk = 0; kk < K; kk += 32) {
        v16h a0 = load_ab16x32(A + (size_t)m0 * K + kk, K, lane);
        v16h a1 = load_ab16x32(A + (size_t)(m0 + 16) * K + kk, K, lane);
#pragma unroll
        for (int j = 0; j < 4; ++j) {
            v16h bt = load_ab16x32(B + (size_t)(n0 + j * 16) * K + kk, K, lane);
            acc[0][j] = __builtin_amdgcn_wmma_f32_16x16x32_f16(
                false, a0, false, bt, (short)0, acc[0][j], false, false);
            acc[1][j] = __builtin_amdgcn_wmma_f32_16x16x32_f16(
                false, a1, false, bt, (short)0, acc[1][j], false, false);
        }
    }

    // C-tile layout: lane holds (M = r + 8*half, N = lane&15) in VGPR r.
    const int col = lane & 15, half = lane >> 4;
#pragma unroll
    for (int i = 0; i < 2; ++i)
#pragma unroll
        for (int j = 0; j < 4; ++j)
#pragma unroll
            for (int r = 0; r < 8; ++r) {
                int m = m0 + i * 16 + r + 8 * half;
                int n = n0 + j * 16 + col;
                C[(size_t)m * N + n] = acc[i][j][r];
            }
}

// ---------------------------------------------------------------------------
// RoPE + head-major repack:
//   qkv f32 [b,s,3,h,64] ->
//     Qh,Kh f16 [h*B+b][s][64]  (rotated)
//     Vt    f16 [h*B+b][64][S]  (transposed so PV's B-operand is row-major-K)
// ---------------------------------------------------------------------------
__global__ __launch_bounds__(256)
void mha_rope_split(const float* __restrict__ qkv, _Float16* __restrict__ Qh,
                    _Float16* __restrict__ Kh, _Float16* __restrict__ Vt) {
    int idx = blockIdx.x * 256 + threadIdx.x;
    int j = idx & 31;  int t = idx >> 5;
    int h = t & 15;    t >>= 4;
    int s = t & (SS - 1);
    int b = t >> 11;

    // angle = s / theta^(2j/64), ln(10000) = 9.2103403719...
    float invf = __expf(-(float)(2 * j) * (9.210340371976184f / 64.0f));
    float ang  = (float)s * invf;
    float sn, cs;
    __sincosf(ang, &sn, &cs);

    const size_t tok = (size_t)(b * SS + s);
    const float* q = qkv + ((tok * 3 + 0) * HH + h) * DH;
    const float* k = qkv + ((tok * 3 + 1) * HH + h) * DH;
    const float* v = qkv + ((tok * 3 + 2) * HH + h) * DH;

    float qe = q[2 * j], qo = q[2 * j + 1];
    float ke = k[2 * j], ko = k[2 * j + 1];

    const size_t hb = (size_t)(h * BB + b);
    const size_t ro = (hb * SS + s) * DH;
    Qh[ro + 2 * j]     = (_Float16)(qe * cs - qo * sn);
    Qh[ro + 2 * j + 1] = (_Float16)(qe * sn + qo * cs);
    Kh[ro + 2 * j]     = (_Float16)(ke * cs - ko * sn);
    Kh[ro + 2 * j + 1] = (_Float16)(ke * sn + ko * cs);

    Vt[(hb * DH + 2 * j) * SS + s]     = (_Float16)v[2 * j];
    Vt[(hb * DH + 2 * j + 1) * SS + s] = (_Float16)v[2 * j + 1];
}

// ---------------------------------------------------------------------------
// Causal flash attention, one wave per 16-query tile per (head,batch).
// K/V tiles for each 32-key block are staged into double-buffered LDS by the
// CDNA5 async engine (GLOBAL_LOAD_ASYNC_TO_LDS_B128 / s_wait_asynccnt) so the
// next block's data movement overlaps the current block's WMMAs.
// ---------------------------------------------------------------------------
__global__ __launch_bounds__(32)
void mha_flash_attn(const _Float16* __restrict__ Qh, const _Float16* __restrict__ Kh,
                    const _Float16* __restrict__ Vt, _Float16* __restrict__ Ao) {
    __shared__ __align__(16) _Float16 kbuf[2][32 * DH];   // [key][dh]   ld=64
    __shared__ __align__(16) _Float16 vbuf[2][DH * 32];   // [dh][key]   ld=32
    __shared__ __align__(16) _Float16 pshare[16 * 32];

    const int lane = threadIdx.x;
    const int col = lane & 15, half = lane >> 4;
    const int qt = blockIdx.x;            // query tile 0..S/16-1
    const int hb = blockIdx.y;            // h*B + b
    const int b  = hb & (BB - 1);
    const int h  = hb >> 1;
    const size_t qkbase = (size_t)hb * SS * DH;
    const _Float16* Vbase = Vt + (size_t)hb * DH * SS;
    const int q0 = qt * 16;

    v16h a0 = load_ab16x32(Qh + qkbase + (size_t)q0 * DH + 0,  DH, lane);
    v16h a1 = load_ab16x32(Qh + qkbase + (size_t)q0 * DH + 32, DH, lane);

    v8f o[4];
#pragma unroll
    for (int c = 0; c < 4; ++c) o[c] = {};
    float m[8], l[8];
#pragma unroll
    for (int r = 0; r < 8; ++r) { m[r] = -INFINITY; l[r] = 0.0f; }

    const float scale = 0.125f;           // 1/sqrt(64)
    const int nkb = (q0 + 16 + 31) >> 5;  // 32-key blocks covering keys 0..q0+15

    // ---- async stage of one 32-key K/V block into LDS buffer (16 x b128) --
    auto stage = [&](int kb) {
        const int buf = kb & 1;
        const int k0  = kb * 32;
        const _Float16* gk = Kh + qkbase + (size_t)k0 * DH;   // 4KB contiguous
        _Float16* lk = kbuf[buf];
#pragma unroll
        for (int i = 0; i < 8; ++i)
            async_copy16(gk + (i * 32 + lane) * 8, lk + (i * 32 + lane) * 8);
        const _Float16* gv = Vbase + k0;                      // 64 rows x 64B
        _Float16* lv = vbuf[buf];
        const int vrow = lane >> 2;          // 0..7 within 8-row group
        const int vc8  = (lane & 3) * 8;     // f16 offset within 32-key row
#pragma unroll
        for (int i = 0; i < 8; ++i)
            async_copy16(gv + (size_t)(i * 8 + vrow) * SS + vc8,
                         lv + (i * 8 + vrow) * 32 + vc8);
    };

    stage(0);

    for (int kb = 0; kb < nkb; ++kb) {
        const int k0 = kb * 32;
        const bool more = (kb + 1 < nkb);
        if (more) stage(kb + 1);
        wait_async_le(more ? 16 : 0);   // async loads complete in order

        const _Float16* kl = kbuf[kb & 1];
        const _Float16* vl = vbuf[kb & 1];

        // --- scores: two 16x16 tiles over 32 keys, contraction dh=64 -------
        v8f s0 = {}, s1 = {};
        {
            v16h b0 = load_ab16x32(kl + 0 * DH + 0,   DH, lane);
            v16h b1 = load_ab16x32(kl + 0 * DH + 32,  DH, lane);
            s0 = __builtin_amdgcn_wmma_f32_16x16x32_f16(false, a0, false, b0, (short)0, s0, false, false);
            s0 = __builtin_amdgcn_wmma_f32_16x16x32_f16(false, a1, false, b1, (short)0, s0, false, false);
            v16h b2 = load_ab16x32(kl + 16 * DH + 0,  DH, lane);
            v16h b3 = load_ab16x32(kl + 16 * DH + 32, DH, lane);
            s1 = __builtin_amdgcn_wmma_f32_16x16x32_f16(false, a0, false, b2, (short)0, s1, false, false);
            s1 = __builtin_amdgcn_wmma_f32_16x16x32_f16(false, a1, false, b3, (short)0, s1, false, false);
        }

        // --- causal mask + online softmax (per C-tile row r) ---------------
        float p0[8], p1[8];
#pragma unroll
        for (int r = 0; r < 8; ++r) {
            const int qrow = q0 + r + 8 * half;
            float x0 = (k0 + col      <= qrow) ? s0[r] * scale : -INFINITY;
            float x1 = (k0 + 16 + col <= qrow) ? s1[r] * scale : -INFINITY;
            float mx = fmaxf(x0, x1);
#pragma unroll
            for (int off = 8; off; off >>= 1) mx = fmaxf(mx, __shfl_xor(mx, off));
            const float mnew  = fmaxf(m[r], mx);
            const float alpha = __expf(m[r] - mnew);
            p0[r] = __expf(x0 - mnew);
            p1[r] = __expf(x1 - mnew);
            float sum = p0[r] + p1[r];
#pragma unroll
            for (int off = 8; off; off >>= 1) sum += __shfl_xor(sum, off);
            l[r] = l[r] * alpha + sum;
            m[r] = mnew;
#pragma unroll
            for (int c = 0; c < 4; ++c) o[c][r] *= alpha;
        }

        // --- P (C-layout) -> A-layout through LDS ---------------------------
#pragma unroll
        for (int r = 0; r < 8; ++r) {
            pshare[(r + 8 * half) * 32 + col]      = (_Float16)p0[r];
            pshare[(r + 8 * half) * 32 + 16 + col] = (_Float16)p1[r];
        }
        asm volatile("s_wait_dscnt 0" ::: "memory");   // CDNA5 split DS counter
        v16h ap = load_ab16x32(pshare, 32, lane);
        asm volatile("s_wait_dscnt 0" ::: "memory");   // reads done before next stores

        // --- O += P(16x32) * V(32keys x 64dh), from staged LDS -------------
#pragma unroll
        for (int c = 0; c < 4; ++c) {
            v16h bv = load_ab16x32(vl + (c * 16) * 32, 32, lane);
            o[c] = __builtin_amdgcn_wmma_f32_16x16x32_f16(false, ap, false, bv, (short)0, o[c], false, false);
        }
    }

    // --- finalize and store head output into [b*S+q][h*64 + d] -------------
#pragma unroll
    for (int r = 0; r < 8; ++r) {
        const float inv = 1.0f / l[r];
#pragma unroll
        for (int c = 0; c < 4; ++c) o[c][r] *= inv;
    }
#pragma unroll
    for (int c = 0; c < 4; ++c)
#pragma unroll
        for (int r = 0; r < 8; ++r) {
            const int qrow = q0 + r + 8 * half;
            Ao[((size_t)b * SS + qrow) * DD + h * DH + c * 16 + col] = (_Float16)o[c][r];
        }
}

// ---------------------------------------------------------------------------
extern "C" void kernel_launch(void* const* d_in, const int* in_sizes, int n_in,
                              void* d_out, int out_size, void* d_ws, size_t ws_size,
                              hipStream_t stream) {
    const float* X    = (const float*)d_in[0];   // [B,S,D]
    const float* Wqkv = (const float*)d_in[1];   // [3D,D]
    const float* Wo   = (const float*)d_in[2];   // [D,D]
    float* out = (float*)d_out;                  // [B,S,D]

    char* ws = (char*)d_ws;
    size_t off = 0;
    auto alloc = [&](size_t bytes) -> char* {
        char* p = ws + off;
        off += (bytes + 255) & ~(size_t)255;
        return p;
    };
    const size_t M = (size_t)BB * SS;  // 4096 tokens
    _Float16* Xh  = (_Float16*)alloc(M * DD * 2);
    _Float16* Wqh = (_Float16*)alloc((size_t)3 * DD * DD * 2);
    _Float16* Woh = (_Float16*)alloc((size_t)DD * DD * 2);
    float*    qkv = (float*)   alloc(M * 3 * DD * 4);
    _Float16* Qh  = (_Float16*)alloc(M * DD * 2);
    _Float16* Kh  = (_Float16*)alloc(M * DD * 2);
    _Float16* Vt  = (_Float16*)alloc(M * DD * 2);
    _Float16* Ao  = (_Float16*)alloc(M * DD * 2);

    const int nX = BB * SS * DD;
    const int nWq = 3 * DD * DD;
    const int nWo = DD * DD;
    mha_cvt_f16<<<(nX + 255) / 256, 256, 0, stream>>>(X, Xh, nX);
    mha_cvt_f16<<<(nWq + 255) / 256, 256, 0, stream>>>(Wqkv, Wqh, nWq);
    mha_cvt_f16<<<(nWo + 255) / 256, 256, 0, stream>>>(Wo, Woh, nWo);

    // qkv[m,o] = sum_d X[m,d] * Wqkv[o,d]
    mha_gemm_tn<<<dim3(3 * DD / 64, (int)M / 32), 32, 0, stream>>>(
        Xh, Wqh, qkv, (int)M, 3 * DD, DD);

    mha_rope_split<<<(BB * SS * HH * 32) / 256, 256, 0, stream>>>(qkv, Qh, Kh, Vt);

    mha_flash_attn<<<dim3(SS / 16, HH * BB), 32, 0, stream>>>(Qh, Kh, Vt, Ao);

    // out[m,o] = sum_d Ao[m,d] * Wo[o,d]
    mha_gemm_tn<<<dim3(DD / 64, (int)M / 32), 32, 0, stream>>>(
        Ao, Woh, out, (int)M, DD, DD);
}
